// HeteroGNNLayer_3616362463347
// MI455X (gfx1250) — matrix operand order repaired and checked by reading, not compiled, not run on hardware
//
#include <hip/hip_runtime.h>
#include <hip/hip_bf16.h>

#define HIDDEN 128
#define K2H    256   // 2*HIDDEN

typedef __attribute__((ext_vector_type(16))) __bf16 v16bf;
typedef __attribute__((ext_vector_type(8)))  __bf16 v8bf;
typedef __attribute__((ext_vector_type(8)))  float  v8f;

// ---------------------------------------------------------------------------
// helpers
// ---------------------------------------------------------------------------

// load 8 consecutive f32, scale, convert to bf16 into frag elements [base..base+7]
__device__ __forceinline__ void load8_cvt(const float* __restrict__ p, float scale,
                                          v16bf& frag, int base) {
  float4 f0 = *reinterpret_cast<const float4*>(p);
  float4 f1 = *reinterpret_cast<const float4*>(p + 4);
  frag[base + 0] = (__bf16)(f0.x * scale);
  frag[base + 1] = (__bf16)(f0.y * scale);
  frag[base + 2] = (__bf16)(f0.z * scale);
  frag[base + 3] = (__bf16)(f0.w * scale);
  frag[base + 4] = (__bf16)(f1.x * scale);
  frag[base + 5] = (__bf16)(f1.y * scale);
  frag[base + 6] = (__bf16)(f1.z * scale);
  frag[base + 7] = (__bf16)(f1.w * scale);
}

// B fragment: this lane holds 16 consecutive K values of its column
// (lanes 0-15: K=kb..kb+15 ; lanes 16-31: K=kb+16..kb+31, offset folded by caller)
__device__ __forceinline__ v16bf load_bfrag(const __bf16* __restrict__ p) {
  v8bf lo = *reinterpret_cast<const v8bf*>(p);
  v8bf hi = *reinterpret_cast<const v8bf*>(p + 8);
  v16bf f;
#pragma unroll
  for (int i = 0; i < 8; ++i) { f[i] = lo[i]; f[i + 8] = hi[i]; }
  return f;
}

// A fragment build (contiguous K order, same as B fragment):
//   elements 0..7  : K = kb + half*16 .. +7       (VGPR 0-3)
//   elements 8..15 : K = kb + half*16 + 8 .. +7   (VGPR 4-7)
// Row source: o < 128 -> rowA[o] ; else rowB[o-128]*scaleB
__device__ __forceinline__ void build_afrag(const float* __restrict__ rowA,
                                            const float* __restrict__ rowB,
                                            float scaleB, int kb, int half,
                                            v16bf& frag) {
  const int o0 = kb + half * 16;
  const int o1 = o0 + 8;
  if (o0 < HIDDEN) load8_cvt(rowA + o0, 1.0f, frag, 0);
  else             load8_cvt(rowB + (o0 - HIDDEN), scaleB, frag, 0);
  if (o1 < HIDDEN) load8_cvt(rowA + o1, 1.0f, frag, 8);
  else             load8_cvt(rowB + (o1 - HIDDEN), scaleB, frag, 8);
}

// ---------------------------------------------------------------------------
// prep: zero scratch, convert/transpose weights to bf16 [cols][K]
// ---------------------------------------------------------------------------
__global__ void zero_kernel(float* __restrict__ p, size_t n) {
  size_t i = (size_t)blockIdx.x * blockDim.x + threadIdx.x;
  const size_t stride = (size_t)gridDim.x * blockDim.x;
  for (; i < n; i += stride) p[i] = 0.0f;
}

__global__ void prep_weights_kernel(const float* __restrict__ Wmsg,   // [3][256][128]
                                    const float* __restrict__ Wupd,   // [256][128]
                                    __bf16* __restrict__ Wmsg_t,      // [3][128][256]
                                    __bf16* __restrict__ Wupd_t) {    // [128][256]
  const int idx = blockIdx.x * blockDim.x + threadIdx.x;
  const int nm = 3 * HIDDEN * K2H;
  if (idx < nm) {
    const int t   = idx / (HIDDEN * K2H);
    const int rem = idx % (HIDDEN * K2H);
    const int n   = rem / K2H;
    const int k   = rem % K2H;
    Wmsg_t[idx] = (__bf16)Wmsg[((size_t)t * K2H + k) * HIDDEN + n];
  } else if (idx < nm + HIDDEN * K2H) {
    const int rem = idx - nm;
    const int n   = rem / K2H;
    const int k   = rem % K2H;
    Wupd_t[rem] = (__bf16)Wupd[(size_t)k * HIDDEN + n];
  }
}

// ---------------------------------------------------------------------------
// edge kernel: typed message GEMM (bf16 WMMA) + atomic scatter-add
// one wave handles a tile of 16 edges x 128 outputs
// ---------------------------------------------------------------------------
__global__ void __launch_bounds__(256)
edge_msg_kernel(const float* __restrict__ x,
                const int*   __restrict__ edge_index,  // [2][E]
                const int*   __restrict__ edge_type,   // [E]
                const __bf16* __restrict__ Wt,         // [3][128][256]
                const float* __restrict__ bmsg,        // [3][128]
                float* __restrict__ agg,               // [N][128]
                float* __restrict__ cnt,               // [N]
                int E) {
  const int lane = threadIdx.x & 31;
  const int wave = threadIdx.x >> 5;
  const int tile = blockIdx.x * 8 + wave;
  if (tile * 16 >= E) return;  // wave-uniform exit: EXEC all-ones inside

  const int r    = lane & 15;
  const int half = lane >> 4;
  const int e    = tile * 16 + r;

  const int src = edge_index[e];
  const int dst = edge_index[E + e];
  const int typ = edge_type[e];

  if (half == 0) atomicAdd(&cnt[dst], 1.0f);  // one per edge row

  // A fragments for all 8 K-steps (rows = x[src] || x[dst]), held in VGPRs
  const float* xsrc = x + (size_t)src * HIDDEN;
  const float* xdst = x + (size_t)dst * HIDDEN;
  v16bf a[8];
#pragma unroll
  for (int k = 0; k < 8; ++k)
    build_afrag(xsrc, xdst, 1.0f, k * 32, half, a[k]);

#pragma unroll 1
  for (int jt = 0; jt < 8; ++jt) {
    v8f acc0 = {}, acc1 = {}, acc2 = {};
    const int col = jt * 16 + r;  // lane's output column (N = lane&15 in tile)
    const __bf16* w0 = Wt + ((size_t)0 * HIDDEN + col) * K2H;
    const __bf16* w1 = Wt + ((size_t)1 * HIDDEN + col) * K2H;
    const __bf16* w2 = Wt + ((size_t)2 * HIDDEN + col) * K2H;
#pragma unroll
    for (int k = 0; k < 8; ++k) {
      const int ko = k * 32 + half * 16;  // B: 16 consecutive K per lane
      v16bf b0 = load_bfrag(w0 + ko);
      v16bf b1 = load_bfrag(w1 + ko);
      v16bf b2 = load_bfrag(w2 + ko);
      acc0 = __builtin_amdgcn_wmma_f32_16x16x32_bf16(false, a[k], false, b0, (short)0, acc0, false, false);
      acc1 = __builtin_amdgcn_wmma_f32_16x16x32_bf16(false, a[k], false, b1, (short)0, acc1, false, false);
      acc2 = __builtin_amdgcn_wmma_f32_16x16x32_bf16(false, a[k], false, b2, (short)0, acc2, false, false);
    }
    // C layout: lane covers N=col, rows M = half*8 + v. Select per-row type,
    // add bias, scatter to agg (L2-resident) with f32 atomics.
#pragma unroll
    for (int v = 0; v < 8; ++v) {
      const int row = half * 8 + v;
      const int t_v = __shfl(typ, row, 32);
      const int d_v = __shfl(dst, row, 32);
      float val = (t_v == 0) ? acc0[v] : (t_v == 1) ? acc1[v] : acc2[v];
      val += bmsg[t_v * HIDDEN + col];
      atomicAdd(&agg[(size_t)d_v * HIDDEN + col], val);
    }
  }
}

// ---------------------------------------------------------------------------
// node kernel: update GEMM (bf16 WMMA) + LayerNorm + GELU + residual
// one wave handles 16 nodes x 128 outputs; h tile staged in padded LDS
// ---------------------------------------------------------------------------
#define LDS_STRIDE 132  // 128 + 4 pad: distinct banks per row

__global__ void __launch_bounds__(128)
node_update_kernel(const float* __restrict__ x,
                   const float* __restrict__ agg,
                   const float* __restrict__ cnt,
                   const __bf16* __restrict__ Wt,     // [128][256]
                   const float* __restrict__ bupd,    // [128]
                   const float* __restrict__ gamma,   // [128]
                   const float* __restrict__ beta,    // [128]
                   float* __restrict__ out, int N) {
  __shared__ float hbuf[4][16 * LDS_STRIDE];
  const int lane = threadIdx.x & 31;
  const int wave = threadIdx.x >> 5;
  const int tile = blockIdx.x * 4 + wave;
  if (tile * 16 >= N) return;  // wave-uniform

  const int r    = lane & 15;
  const int half = lane >> 4;
  const int node = tile * 16 + r;

  const float inv = 1.0f / fmaxf(cnt[node], 1.0f);  // count-normalize (>=1)
  const float* xr = x   + (size_t)node * HIDDEN;
  const float* ar = agg + (size_t)node * HIDDEN;

  v16bf a[8];
#pragma unroll
  for (int k = 0; k < 8; ++k)
    build_afrag(xr, ar, inv, k * 32, half, a[k]);

  float* h = &hbuf[wave][0];
#pragma unroll 1
  for (int jt = 0; jt < 8; ++jt) {
    v8f acc = {};
    const int col = jt * 16 + r;
    const __bf16* w = Wt + (size_t)col * K2H;
#pragma unroll
    for (int k = 0; k < 8; ++k) {
      v16bf b = load_bfrag(w + k * 32 + half * 16);
      acc = __builtin_amdgcn_wmma_f32_16x16x32_bf16(false, a[k], false, b, (short)0, acc, false, false);
    }
    const float bias = bupd[col];
#pragma unroll
    for (int v = 0; v < 8; ++v) {
      const int row = half * 8 + v;
      h[row * LDS_STRIDE + col] = acc[v] + bias;
    }
  }

  // per-wave LDS traffic only: order with dscnt, no workgroup barrier needed
  asm volatile("s_wait_dscnt 0" ::: "memory");

  // LayerNorm + exact GELU + residual.
  // lane handles row r, columns [half*64, half*64+64); reduce pair via xor-16.
  const int cb = half * 64;
  const float* hr = h + r * LDS_STRIDE;
  float s1 = 0.0f, s2 = 0.0f;
#pragma unroll
  for (int i = 0; i < 64; ++i) {
    const float v = hr[cb + i];
    s1 += v;
    s2 += v * v;
  }
  s1 += __shfl_xor(s1, 16, 32);
  s2 += __shfl_xor(s2, 16, 32);
  const float mu   = s1 * (1.0f / 128.0f);
  const float var  = s2 * (1.0f / 128.0f) - mu * mu;
  const float rstd = rsqrtf(var + 1e-5f);

  const float* xrow = x   + (size_t)node * HIDDEN;
  float*       orow = out + (size_t)node * HIDDEN;
#pragma unroll
  for (int i = 0; i < 64; ++i) {
    const int c = cb + i;
    const float hn = (hr[c] - mu) * rstd * gamma[c] + beta[c];
    const float g  = 0.5f * hn * (1.0f + erff(hn * 0.70710678118654752f));
    orow[c] = xrow[c] + g;
  }
}

// ---------------------------------------------------------------------------
// launch
// ---------------------------------------------------------------------------
extern "C" void kernel_launch(void* const* d_in, const int* in_sizes, int n_in,
                              void* d_out, int out_size, void* d_ws, size_t ws_size,
                              hipStream_t stream) {
  const float* x     = (const float*)d_in[0];
  const int*   ei    = (const int*)d_in[1];   // [2][E]
  const int*   et    = (const int*)d_in[2];   // [E]
  const float* Wmsg  = (const float*)d_in[3]; // [3][256][128]
  const float* bmsg  = (const float*)d_in[4]; // [3][128]
  const float* Wupd  = (const float*)d_in[5]; // [256][128]
  const float* bupd  = (const float*)d_in[6]; // [128]
  const float* gamma = (const float*)d_in[7]; // [128]
  const float* beta  = (const float*)d_in[8]; // [128]

  const int N = in_sizes[0] / HIDDEN;  // 50000
  const int E = in_sizes[2];           // 600000

  // workspace layout: agg [N*128 f32] | cnt [N f32] | Wmsg_t bf16 | Wupd_t bf16
  char* ws = (char*)d_ws;
  float* agg = (float*)ws;
  size_t off = (size_t)N * HIDDEN * sizeof(float);
  float* cnt = (float*)(ws + off);
  off += (size_t)N * sizeof(float);
  off = (off + 255) & ~(size_t)255;  // 16B+ alignment for b128 loads
  __bf16* Wmsg_t = (__bf16*)(ws + off);
  off += (size_t)3 * HIDDEN * K2H * sizeof(__bf16);
  __bf16* Wupd_t = (__bf16*)(ws + off);

  // 1) zero agg+cnt (contiguous: N*129 floats); 2) weight convert/transpose
  zero_kernel<<<2048, 256, 0, stream>>>(agg, (size_t)N * (HIDDEN + 1));
  prep_weights_kernel<<<(4 * HIDDEN * K2H + 255) / 256, 256, 0, stream>>>(
      Wmsg, Wupd, Wmsg_t, Wupd_t);

  // 3) edge messages + scatter (E/16 tiles, 8 waves/block)
  const int etiles = E / 16;
  edge_msg_kernel<<<(etiles + 7) / 8, 256, 0, stream>>>(
      x, ei, et, Wmsg_t, bmsg, agg, cnt, E);

  // 4) node update + LN + GELU + residual (N/16 tiles, 4 waves/block)
  const int ntiles = (N + 15) / 16;
  node_update_kernel<<<(ntiles + 3) / 4, 128, 0, stream>>>(
      x, agg, cnt, Wupd_t, bupd, gamma, beta, (float*)d_out, N);
}
